// GRU_21423296873043
// MI455X (gfx1250) — matrix operand order repaired and checked by reading, not compiled, not run on hardware
//
#include <hip/hip_runtime.h>

typedef unsigned short u16;
typedef __attribute__((ext_vector_type(16))) __bf16 v16bf;
typedef __attribute__((ext_vector_type(8)))  __bf16 v8bf;
typedef __attribute__((ext_vector_type(8)))  float  v8f;

#define B_      4
#define S_      2048
#define DM_     2048
#define NH_     32
#define D_      64
#define G_      8
#define C1_     1536            // 3*ISS
#define C2_     6144            // 3*NH*D
#define M_      8192            // B*S
#define KC_     64              // K staging chunk
#define FACTOR_ 1.414213562373095f
#define EPS_    1e-6f

__device__ __forceinline__ u16 f32_to_bf16(float f) {
    unsigned u = __float_as_uint(f);
    u += 0x7FFFu + ((u >> 16) & 1u);   // round-to-nearest-even
    return (u16)(u >> 16);
}

// 32-byte global -> LDS async copy (CDNA5 async path, tracked by ASYNCcnt).
// INST_OFFSET is added to both the LDS and the global address.
__device__ __forceinline__ void async_copy32(unsigned lds, const u16* g) {
    unsigned long long ga = (unsigned long long)g;
    asm volatile("global_load_async_to_lds_b128 %0, %1, off\n\t"
                 "global_load_async_to_lds_b128 %0, %1, off offset:16"
                 :: "v"(lds), "v"(ga) : "memory");
}

// ---------------------------------------------------------------------------
// elementwise f32 -> bf16
__global__ __launch_bounds__(256)
void cvt_bf16(const float* __restrict__ in, u16* __restrict__ out, long n) {
    long i = (long)blockIdx.x * 256 + threadIdx.x;
    if (i < n) out[i] = f32_to_bf16(in[i]);
}

// (bat, I, O) f32 -> (bat, O, I) bf16   (makes weights (N,K) row-major)
__global__ __launch_bounds__(256)
void cvt_bf16_T(const float* __restrict__ in, u16* __restrict__ out, int I, int O) {
    long idx = (long)blockIdx.x * 256 + threadIdx.x;
    long total = (long)I * O;
    if (idx >= total) return;
    long bat = blockIdx.y;
    int o = (int)(idx % O);
    int i = (int)(idx / O);
    out[bat * total + (long)o * I + i] = f32_to_bf16(in[bat * total + (long)i * O + o]);
}

// ---------------------------------------------------------------------------
// C[M,N] f32 = scale * (A[M,K]bf16 @ B[N,K]bf16^T) + bias  (batched)
// grid: (N/64, M/128, batch); block: 256 (8 waves); wave -> 16 rows x 64 cols.
// B panel (64 rows x KC) staged in LDS via async loads, double buffered.
__global__ __launch_bounds__(256)
void gemm_bf16_tn(const u16* __restrict__ A, long lda, long sA,
                  const u16* __restrict__ Bm, long ldb, long sB,
                  float* __restrict__ C, long ldc, long sC,
                  const float* __restrict__ bias, float scale, int K)
{
    __shared__ u16 Bs[2][64 * KC_];

    const int tid  = threadIdx.x;
    const int lane = tid & 31;
    const int wave = tid >> 5;
    const long m0 = (long)blockIdx.y * 128 + wave * 16;
    const long n0 = (long)blockIdx.x * 64;
    A  += (long)blockIdx.z * sA;
    Bm += (long)blockIdx.z * sB;
    C  += (long)blockIdx.z * sC;

    // async staging geometry: thread t copies 32B: row = t/4, 32B segment = t%4
    const int brow = tid >> 2;
    const int bseg = (tid & 3) * 16;                 // element offset within chunk
    const u16* gb = Bm + (n0 + brow) * ldb + bseg;   // + kc at issue time
    const unsigned lds0 =
        (unsigned)(unsigned long long)&Bs[0][brow * KC_ + bseg];
    const unsigned lds1 =
        (unsigned)(unsigned long long)&Bs[1][brow * KC_ + bseg];

    // wave32 WMMA striping used here, contiguous k for A and B (16-bit A 16x32, B 32x16)
    const int mrow  = lane & 15;
    const int koffA = (lane >> 4) * 16;   // low lanes K0..15, high lanes K16..31
    const int nloc  = lane & 15;
    const int koffB = (lane >> 4) * 16;   // low lanes K0..15, high lanes K16..31
    const u16* pa = A + (m0 + mrow) * lda;

    v8f acc0 = {0.f,0.f,0.f,0.f,0.f,0.f,0.f,0.f};
    v8f acc1 = acc0, acc2 = acc0, acc3 = acc0;

    const int nch = K / KC_;
    async_copy32(lds0, gb);                          // chunk 0 -> buf 0
    for (int c = 0; c < nch; ++c) {
        const int kc = c * KC_;
        if (c + 1 < nch) {
            async_copy32((c & 1) ? lds0 : lds1, gb + kc + KC_);
            asm volatile("s_wait_asynccnt 0x2" ::: "memory");  // chunk c landed
        } else {
            asm volatile("s_wait_asynccnt 0x0" ::: "memory");
        }
        __syncthreads();

        const u16* bp = &Bs[c & 1][0];
        __builtin_prefetch(pa + kc + 2 * KC_, 0, 1);           // global_prefetch_b8 for A
        #pragma unroll
        for (int kk = 0; kk < KC_; kk += 32) {
            v8bf alo = *reinterpret_cast<const v8bf*>(pa + kc + kk + koffA);
            v8bf ahi = *reinterpret_cast<const v8bf*>(pa + kc + kk + 8 + koffA);
            v16bf af = __builtin_shufflevector(alo, ahi, 0,1,2,3,4,5,6,7,
                                                         8,9,10,11,12,13,14,15);
            v16bf b0 = *reinterpret_cast<const v16bf*>(bp + ( 0 + nloc) * KC_ + kk + koffB);
            v16bf b1 = *reinterpret_cast<const v16bf*>(bp + (16 + nloc) * KC_ + kk + koffB);
            v16bf b2 = *reinterpret_cast<const v16bf*>(bp + (32 + nloc) * KC_ + kk + koffB);
            v16bf b3 = *reinterpret_cast<const v16bf*>(bp + (48 + nloc) * KC_ + kk + koffB);
            acc0 = __builtin_amdgcn_wmma_f32_16x16x32_bf16(false, af, false, b0, (short)0, acc0, false, false);
            acc1 = __builtin_amdgcn_wmma_f32_16x16x32_bf16(false, af, false, b1, (short)0, acc1, false, false);
            acc2 = __builtin_amdgcn_wmma_f32_16x16x32_bf16(false, af, false, b2, (short)0, acc2, false, false);
            acc3 = __builtin_amdgcn_wmma_f32_16x16x32_bf16(false, af, false, b3, (short)0, acc3, false, false);
        }
        __syncthreads();   // all waves done reading before this buffer is re-filled
    }

    // C/D layout: VGPR v, lanes0-15 -> M=v, lanes16-31 -> M=8+v; N=lane%16
    const long rbase = m0 + (lane >> 4) * 8;
    const long cbase = n0 + nloc;
    #pragma unroll
    for (int v = 0; v < 8; ++v) {
        float* cr = C + (rbase + v) * ldc + cbase;
        float b0v = bias ? bias[cbase +  0] : 0.f;
        float b1v = bias ? bias[cbase + 16] : 0.f;
        float b2v = bias ? bias[cbase + 32] : 0.f;
        float b3v = bias ? bias[cbase + 48] : 0.f;
        cr[ 0] = acc0[v] * scale + b0v;
        cr[16] = acc1[v] * scale + b1v;
        cr[32] = acc2[v] * scale + b2v;
        cr[48] = acc3[v] * scale + b3v;
    }
}

// ---------------------------------------------------------------------------
// rmsnorm over last dim, f32 in -> bf16 out. One block per row.
__global__ __launch_bounds__(256)
void rmsnorm_bf16(const float* __restrict__ X, const float* __restrict__ w,
                  u16* __restrict__ Y, int C)
{
    __shared__ float red[256];
    const long row = blockIdx.x;
    const float* x = X + row * (long)C;
    float ss = 0.f;
    for (int i = threadIdx.x; i < C; i += 256) { float v = x[i]; ss += v * v; }
    red[threadIdx.x] = ss;
    __syncthreads();
    for (int s2 = 128; s2 > 0; s2 >>= 1) {
        if (threadIdx.x < s2) red[threadIdx.x] += red[threadIdx.x + s2];
        __syncthreads();
    }
    const float sc = rsqrtf(red[0] * (1.0f / (float)C) + EPS_);
    u16* y = Y + row * (long)C;
    for (int i = threadIdx.x; i < C; i += 256)
        y[i] = f32_to_bf16(x[i] * sc * w[i]);
}

// ---------------------------------------------------------------------------
// Sequential gated scan. One wave32 block per (b, head). Lane owns channels
// e and e+32; all 3 gate weight columns live in VGPRs; LDS broadcasts state.
// gates: (b, s, 6144) f32 pre-scaled by FACTOR; sw: (96,64,64) f32 raw.
// ybf: (b, s, 2048) bf16 output (feeds out-head GEMM).
__global__ __launch_bounds__(32)
void gru_scan(const float* __restrict__ gates, const float* __restrict__ sw,
              u16* __restrict__ ybf)
{
    const int n  = blockIdx.x & 31;
    const int b  = blockIdx.x >> 5;
    const int e0 = threadIdx.x;
    const int e1 = e0 + 32;

    __shared__ float hsh[64];
    __shared__ float rhsh[64];

    float wi0[64], wi1[64], wf0[64], wf1[64], wr0[64], wr1[64];
    const long bi = (long)(0 * NH_ + n) * 4096;   // wi
    const long bf = (long)(1 * NH_ + n) * 4096;   // wf
    const long br = (long)(2 * NH_ + n) * 4096;   // wr
    #pragma unroll
    for (int d = 0; d < 64; ++d) {
        wi0[d] = sw[bi + d * 64 + e0] * FACTOR_;  wi1[d] = sw[bi + d * 64 + e1] * FACTOR_;
        wf0[d] = sw[bf + d * 64 + e0] * FACTOR_;  wf1[d] = sw[bf + d * 64 + e1] * FACTOR_;
        wr0[d] = sw[br + d * 64 + e0] * FACTOR_;  wr1[d] = sw[br + d * 64 + e1] * FACTOR_;
    }

    float h0 = 0.f, h1 = 0.f;
    hsh[e0] = 0.f; hsh[e1] = 0.f;
    __syncthreads();

    for (int s = 0; s < S_; ++s) {
        const long base = ((long)b * S_ + s) * (long)C2_ + n * 64;
        float it0 = gates[base + e0],        it1 = gates[base + e1];
        float ft0 = gates[base + 2048 + e0], ft1 = gates[base + 2048 + e1];
        float rt0 = gates[base + 4096 + e0], rt1 = gates[base + 4096 + e1];

        float af0 = ft0, af1 = ft1, ar0 = rt0, ar1 = rt1;
        #pragma unroll
        for (int d = 0; d < 64; ++d) {
            float hv = hsh[d];
            af0 += hv * wf0[d]; af1 += hv * wf1[d];
            ar0 += hv * wr0[d]; ar1 += hv * wr1[d];
        }
        float f0 = 1.f / (1.f + __expf(-af0));
        float f1 = 1.f / (1.f + __expf(-af1));
        float r0 = 1.f / (1.f + __expf(-ar0));
        float r1 = 1.f / (1.f + __expf(-ar1));

        rhsh[e0] = r0 * h0; rhsh[e1] = r1 * h1;
        __syncthreads();

        float an0 = it0, an1 = it1;
        #pragma unroll
        for (int d = 0; d < 64; ++d) {
            float rv = rhsh[d];
            an0 += rv * wi0[d]; an1 += rv * wi1[d];
        }
        float nn0 = tanhf(an0), nn1 = tanhf(an1);
        h0 = f0 * h0 + (1.f - f0) * nn0;
        h1 = f1 * h1 + (1.f - f1) * nn1;

        const long yb = ((long)b * S_ + s) * (long)(NH_ * D_) + n * 64;
        ybf[yb + e0] = f32_to_bf16(h0);
        ybf[yb + e1] = f32_to_bf16(h1);

        hsh[e0] = h0; hsh[e1] = h1;
        __syncthreads();
    }
}

// ---------------------------------------------------------------------------
extern "C" void kernel_launch(void* const* d_in, const int* in_sizes, int n_in,
                              void* d_out, int out_size, void* d_ws, size_t ws_size,
                              hipStream_t stream) {
    (void)in_sizes; (void)n_in; (void)out_size; (void)ws_size;
    const float* x        = (const float*)d_in[0];
    const float* W_in     = (const float*)d_in[1];
    const float* b_in     = (const float*)d_in[2];
    const float* rms1_w   = (const float*)d_in[3];
    const float* W_head   = (const float*)d_in[4];
    const float* state_w  = (const float*)d_in[5];
    const float* W_outh   = (const float*)d_in[6];
    const float* rms2_w   = (const float*)d_in[7];
    const float* W_out    = (const float*)d_in[8];
    float* out = (float*)d_out;

    char* ws = (char*)d_ws;
    size_t off = 0;
    auto alloc = [&](size_t bytes) -> void* {
        void* p = ws + off;
        off = (off + bytes + 255) & ~(size_t)255;
        return p;
    };
    u16*   x_bf     = (u16*)  alloc((size_t)M_ * DM_ * 2);      // 33.5 MB
    u16*   Win_bf   = (u16*)  alloc((size_t)C1_ * DM_ * 2);     //  6.3 MB
    u16*   WheadT   = (u16*)  alloc((size_t)G_ * 768 * 192 * 2);//  2.4 MB
    u16*   WouthT   = (u16*)  alloc((size_t)NH_ * 64 * 64 * 2); //  0.3 MB
    u16*   Wout_bf  = (u16*)  alloc((size_t)DM_ * DM_ * 2);     //  8.4 MB
    float* h1       = (float*)alloc((size_t)M_ * C1_ * 4);      // 50.3 MB
    u16*   h1n_bf   = (u16*)  alloc((size_t)M_ * C1_ * 2);      // 25.2 MB
    float* gates    = (float*)alloc((size_t)M_ * C2_ * 4);      // 201  MB
    u16*   y_bf     = (u16*)  alloc((size_t)M_ * DM_ * 2);      // 33.5 MB
    float* z        = (float*)alloc((size_t)M_ * DM_ * 4);      // 67.1 MB
    u16*   z_bf     = (u16*)  alloc((size_t)M_ * DM_ * 2);      // 33.5 MB

    // --- one-time operand conversion (bf16, weights transposed to (N,K)) ---
    {
        long n = (long)M_ * DM_;
        cvt_bf16<<<dim3((unsigned)((n + 255) / 256)), 256, 0, stream>>>(x, x_bf, n);
        n = (long)C1_ * DM_;
        cvt_bf16<<<dim3((unsigned)((n + 255) / 256)), 256, 0, stream>>>(W_in, Win_bf, n);
        n = (long)DM_ * DM_;
        cvt_bf16<<<dim3((unsigned)((n + 255) / 256)), 256, 0, stream>>>(W_out, Wout_bf, n);
        cvt_bf16_T<<<dim3((192u * 768u + 255) / 256, G_), 256, 0, stream>>>(W_head, WheadT, 192, 768);
        cvt_bf16_T<<<dim3((64u * 64u + 255) / 256, NH_), 256, 0, stream>>>(W_outh, WouthT, 64, 64);
    }

    // --- h1 = x @ W_in^T + b_in ---
    gemm_bf16_tn<<<dim3(C1_ / 64, M_ / 128, 1), 256, 0, stream>>>(
        x_bf, DM_, 0, Win_bf, DM_, 0, h1, C1_, 0, b_in, 1.0f, DM_);

    // --- rmsnorm(h1) -> bf16 ---
    rmsnorm_bf16<<<M_, 256, 0, stream>>>(h1, rms1_w, h1n_bf, C1_);

    // --- grouped head einsum, scaled by FACTOR -> gates (b,s,6144) ---
    gemm_bf16_tn<<<dim3(768 / 64, M_ / 128, G_), 256, 0, stream>>>(
        h1n_bf, C1_, /*sA=*/192,
        WheadT, 192, /*sB=*/(long)768 * 192,
        gates, C2_, /*sC=*/768,
        nullptr, FACTOR_, 192);

    // --- sequential gated recurrence: 128 (b,head) waves ---
    gru_scan<<<B_ * NH_, 32, 0, stream>>>(gates, state_w, y_bf);

    // --- per-head output einsum: z = y @ W_outhead ---
    gemm_bf16_tn<<<dim3(64 / 64, M_ / 128, NH_), 256, 0, stream>>>(
        y_bf, DM_, /*sA=*/64,
        WouthT, 64, /*sB=*/(long)64 * 64,
        z, DM_, /*sC=*/64,
        nullptr, 1.0f, 64);

    // --- rmsnorm(z) -> bf16 ---
    rmsnorm_bf16<<<M_, 256, 0, stream>>>(z, rms2_w, z_bf, DM_);

    // --- out = z @ W_out^T ---
    gemm_bf16_tn<<<dim3(DM_ / 64, M_ / 128, 1), 256, 0, stream>>>(
        z_bf, DM_, 0, Wout_bf, DM_, 0, out, DM_, 0, nullptr, 1.0f, DM_);
}
